// MultiHeadAttention_79748952752414
// MI455X (gfx1250) — matrix operand order repaired and checked by reading, not compiled, not run on hardware
//
#include <hip/hip_runtime.h>
#include <hip/hip_bf16.h>

// ---------------- problem constants ----------------
#define D_MODEL 1024
#define NHEADS  16
#define DK      64
#define BATCH   2
#define SEQ     2048
#define MTOT    (BATCH * SEQ)      // 4096
#define SCALE   0.125f             // 1/sqrt(64)

// ---------------- WMMA types ----------------
typedef __bf16 bf16_t;
typedef bf16_t v16bf __attribute__((ext_vector_type(16)));
typedef float  v8f   __attribute__((ext_vector_type(8)));

union FragBF {                     // one WMMA bf16 A/B fragment (32 bytes / lane)
    v16bf v;
    uint4 q[2];
};

// f32 -> bf16 round-to-nearest-even
static __device__ __forceinline__ unsigned short f2bf(float f) {
    unsigned int u = __float_as_uint(f);
    u += 0x7FFFu + ((u >> 16) & 1u);
    return (unsigned short)(u >> 16);
}

// ---------------- CDNA5 async global->LDS copy (guarded) ----------------
#if __has_builtin(__builtin_amdgcn_global_load_async_to_lds_b128)
#define HAVE_ASYNC_LDS 1
typedef int v4i_ __attribute__((ext_vector_type(4)));
typedef __attribute__((address_space(1))) v4i_ GI4;   // global int4
typedef __attribute__((address_space(3))) v4i_ LI4;   // LDS int4
static __device__ __forceinline__ void async_cp16(const unsigned short* g,
                                                  unsigned short* l) {
    __builtin_amdgcn_global_load_async_to_lds_b128((GI4*)g, (LI4*)l, 0, 0);
}
#else
#define HAVE_ASYNC_LDS 0
#endif

// ---------------- elementwise prep kernels ----------------
__global__ void mha_cast_bf16(const float* __restrict__ src,
                              unsigned short* __restrict__ dst, int n) {
    int i = blockIdx.x * blockDim.x + threadIdx.x;
    if (i < n) dst[i] = f2bf(src[i]);
}

// W (KxN f32, row-major) -> Wt (NxK bf16, row-major)
__global__ void mha_wtrans(const float* __restrict__ W,
                           unsigned short* __restrict__ Wt) {
    int i = blockIdx.x * blockDim.x + threadIdx.x;
    int n = i & (D_MODEL - 1);
    int k = i >> 10;
    Wt[(size_t)n * D_MODEL + k] = f2bf(W[(size_t)k * D_MODEL + n]);
}

// RoPE on (BH, S, DK) f32 -> bf16 (same layout); one thread per even/odd pair
__global__ void mha_rope_cast(const float* __restrict__ src,
                              unsigned short* __restrict__ dst, int npairs) {
    int i = blockIdx.x * blockDim.x + threadIdx.x;
    if (i >= npairs) return;
    int j = i & 31;                  // pair index within head dim (DK/2 = 32)
    int s = (i >> 5) & (SEQ - 1);    // sequence position
    // theta_j = 10000^(-2j/64) = exp(-(2j) * ln(10000)/64)
    float theta = __expf(-0.14391156f * (float)(2 * j));
    float ang = (float)s * theta;
    float sn, cs;
    __sincosf(ang, &sn, &cs);
    float xe = src[2 * i], xo = src[2 * i + 1];
    dst[2 * i]     = f2bf(xe * cs - xo * sn);
    dst[2 * i + 1] = f2bf(xe * sn + xo * cs);
}

// V: (BH, S, DK) f32 -> (BH, DK, S) bf16 (transposed so P*V B-frags are contiguous)
__global__ void mha_vcast_t(const float* __restrict__ src,
                            unsigned short* __restrict__ dst, int total) {
    int i = blockIdx.x * blockDim.x + threadIdx.x;
    if (i >= total) return;
    int d  = i & (DK - 1);
    int s  = (i >> 6) & (SEQ - 1);
    int bh = i >> 17;                // 6 (DK) + 11 (SEQ) bits
    dst[((size_t)bh * DK + d) * SEQ + s] = f2bf(src[i]);
}

// ---------------- WMMA GEMM:  out = A(bf16 MxK) * Wt(bf16 NxK)^T + bias ----------------
// Block 128x128, 8 waves (4x2), wave tile 32x64 (2x4 WMMA accumulators).
// MODE 0: out[m*N + n] (f32, row-major)           -- final projection
// MODE 1: out[((b*H + h)*S + s)*DK + d] (f32)     -- per-head split for attention
template <int MODE>
__global__ __launch_bounds__(256) void mha_gemm_bf16(
    const unsigned short* __restrict__ A,    // M x K bf16 row-major
    const unsigned short* __restrict__ Wt,   // N x K bf16 row-major (pre-transposed W)
    const float* __restrict__ bias,          // N
    float* __restrict__ out, int M, int N, int K) {
    const int tid   = threadIdx.x;
    const int wid   = tid >> 5;
    const int lane  = tid & 31;
    const int lrow  = lane & 15;     // row (A) / column (B,C) owned by this lane
    const int hf    = lane >> 4;     // half-wave selector
    const int waveM = wid & 3, waveN = wid >> 2;     // 4x2 wave grid
    const int m_base = blockIdx.x * 128 + waveM * 32;
    const int n_base = blockIdx.y * 128 + waveN * 64;

    v8f acc[2][4] = {};
    for (int k0 = 0; k0 < K; k0 += 32) {
        FragBF a[2], b[4];
#pragma unroll
        for (int mi = 0; mi < 2; ++mi) {
            const unsigned short* ap =
                A + (size_t)(m_base + mi * 16 + lrow) * K + k0 + hf * 8;
            a[mi].q[0] = *(const uint4*)ap;
            a[mi].q[1] = *(const uint4*)(ap + 16);
            __builtin_prefetch(ap + 32, 0, 1);       // global_prefetch_b8 next K step
        }
#pragma unroll
        for (int ni = 0; ni < 4; ++ni) {
            const unsigned short* bp =
                Wt + (size_t)(n_base + ni * 16 + lrow) * K + k0 + hf * 8;
            b[ni].q[0] = *(const uint4*)bp;
            b[ni].q[1] = *(const uint4*)(bp + 16);
        }
#pragma unroll
        for (int mi = 0; mi < 2; ++mi)
#pragma unroll
            for (int ni = 0; ni < 4; ++ni)
                acc[mi][ni] = __builtin_amdgcn_wmma_f32_16x16x32_bf16(
                    false, a[mi].v, false, b[ni].v, (short)0, acc[mi][ni], false, false);
    }

#pragma unroll
    for (int mi = 0; mi < 2; ++mi)
#pragma unroll
        for (int ni = 0; ni < 4; ++ni) {
            const int col = n_base + ni * 16 + lrow;
            const float bv = bias[col];
#pragma unroll
            for (int r = 0; r < 8; ++r) {
                const int row = m_base + mi * 16 + hf * 8 + r;  // C layout row
                const float v = acc[mi][ni][r] + bv;
                if (MODE == 0) {
                    out[(size_t)row * N + col] = v;
                } else {
                    const int bb = row >> 11, s = row & (SEQ - 1);
                    const int h = col >> 6, d = col & (DK - 1);
                    out[(((size_t)(bb * NHEADS + h) * SEQ + s) << 6) + d] = v;
                }
            }
        }
}

// ---------------- flash attention (causal, online softmax) ----------------
// grid: (SEQ/64, B*H), block: 128 threads (4 waves); wave owns 16 query rows.
__global__ __launch_bounds__(128) void mha_flash_attn(
    const unsigned short* __restrict__ Qh,   // (BH, S, DK) bf16, RoPE'd
    const unsigned short* __restrict__ Kh,   // (BH, S, DK) bf16, RoPE'd
    const unsigned short* __restrict__ Vt,   // (BH, DK, S) bf16
    unsigned short* __restrict__ ctx) {      // (B, S, D_MODEL) bf16
#if HAVE_ASYNC_LDS
    __shared__ unsigned short lds_k[4][32 * DK];   // per-wave 4KB K tile (32 keys x 64)
    __shared__ unsigned short lds_v[4][DK * 32];   // per-wave 4KB V tile (64 d x 32 keys)
#endif
    __shared__ unsigned short lds_p[4][16 * 32];   // per-wave 16x32 P staging

    const int tid  = threadIdx.x;
    const int wid  = tid >> 5;
    const int lane = tid & 31;
    const int lrow = lane & 15;
    const int hf   = lane >> 4;
    const int bh   = blockIdx.y;
    const int q0   = blockIdx.x * 64 + wid * 16;

    const unsigned short* Qb = Qh + (size_t)bh * SEQ * DK;
    const unsigned short* Kb = Kh + (size_t)bh * SEQ * DK;
    const unsigned short* Vb = Vt + (size_t)bh * DK * SEQ;

    // Q A-fragments for d-chunks {0..31},{32..63} kept resident all iterations.
    FragBF aq[2];
#pragma unroll
    for (int c = 0; c < 2; ++c) {
        const unsigned short* p = Qb + (size_t)(q0 + lrow) * DK + c * 32 + hf * 8;
        aq[c].q[0] = *(const uint4*)p;
        aq[c].q[1] = *(const uint4*)(p + 16);
    }

    float m_i[8], l_i[8];
    v8f o[4] = {};
#pragma unroll
    for (int r = 0; r < 8; ++r) { m_i[r] = -3.0e38f; l_i[r] = 0.0f; }

    const int n_kt = ((q0 + 15) >> 5) + 1;           // causal key-tile count
    for (int kt = 0; kt < n_kt; ++kt) {
        const int kt0 = kt * 32;

#if HAVE_ASYNC_LDS
        // ---- coalesced async staging of K (contiguous 4KB) and V (64x64B rows) ----
        unsigned short* lk = lds_k[wid];
        unsigned short* lv = lds_v[wid];
        {
            const unsigned short* gk = Kb + (size_t)kt0 * DK;  // 32 rows x 128B contiguous
#pragma unroll
            for (int i = 0; i < 8; ++i) {
                const int e = (i * 32 + lane) * 8;             // 8 bf16 = 16B per lane
                async_cp16(gk + e, lk + e);
            }
#pragma unroll
            for (int i = 0; i < 8; ++i) {
                const int g   = i * 8 + (lane >> 2);           // V row (d index)
                const int off = (lane & 3) * 8;                // chunk within 32-elem seg
                async_cp16(Vb + (size_t)g * SEQ + kt0 + off, lv + g * 32 + off);
            }
        }
        asm volatile("s_wait_asynccnt 0" ::: "memory");        // per-wave ASYNCcnt
#endif

        // ---- scores S = Q K^T : two 16x16 C-tiles (key cols j*16..) ----
        v8f sc[2] = {};
#pragma unroll
        for (int j = 0; j < 2; ++j)
#pragma unroll
            for (int c = 0; c < 2; ++c) {
                FragBF bk;
#if HAVE_ASYNC_LDS
                const unsigned short* kp =
                    lk + (j * 16 + lrow) * DK + c * 32 + hf * 8;
#else
                const unsigned short* kp =
                    Kb + (size_t)(kt0 + j * 16 + lrow) * DK + c * 32 + hf * 8;
#endif
                bk.q[0] = *(const uint4*)kp;
                bk.q[1] = *(const uint4*)(kp + 16);
                sc[j] = __builtin_amdgcn_wmma_f32_16x16x32_bf16(
                    false, aq[c].v, false, bk.v, (short)0, sc[j], false, false);
            }

        // ---- scale + causal mask + row max (half-wave shfl reduction) ----
        float mx[8];
#pragma unroll
        for (int r = 0; r < 8; ++r) {
            const int s_q = q0 + hf * 8 + r;
            float v0 = sc[0][r] * SCALE;
            float v1 = sc[1][r] * SCALE;
            if (kt0 + lrow > s_q)      v0 = -1.0e9f;
            if (kt0 + 16 + lrow > s_q) v1 = -1.0e9f;
            sc[0][r] = v0; sc[1][r] = v1;
            float m = fmaxf(v0, v1);
            m = fmaxf(m, __shfl_xor(m, 1));
            m = fmaxf(m, __shfl_xor(m, 2));
            m = fmaxf(m, __shfl_xor(m, 4));
            m = fmaxf(m, __shfl_xor(m, 8));
            mx[r] = m;
        }

        // ---- online softmax update ----
        float p0v[8], p1v[8];
#pragma unroll
        for (int r = 0; r < 8; ++r) {
            const float m_new = fmaxf(m_i[r], mx[r]);
            const float alpha = __expf(m_i[r] - m_new);
            const float p0 = __expf(sc[0][r] - m_new);
            const float p1 = __expf(sc[1][r] - m_new);
            float s = p0 + p1;
            s += __shfl_xor(s, 1);
            s += __shfl_xor(s, 2);
            s += __shfl_xor(s, 4);
            s += __shfl_xor(s, 8);
            l_i[r] = l_i[r] * alpha + s;
            m_i[r] = m_new;
#pragma unroll
            for (int di = 0; di < 4; ++di) o[di][r] *= alpha;
            p0v[r] = p0; p1v[r] = p1;
        }

        // ---- C-layout -> A-layout for P via per-wave LDS bounce ----
        unsigned short* lp = lds_p[wid];
#pragma unroll
        for (int r = 0; r < 8; ++r) {
            lp[(hf * 8 + r) * 32 + lrow]      = f2bf(p0v[r]);
            lp[(hf * 8 + r) * 32 + 16 + lrow] = f2bf(p1v[r]);
        }
        asm volatile("s_wait_dscnt 0" ::: "memory");  // wave-local LDS RAW ordering
        FragBF ap;
        {
            const unsigned short* pp = lp + lrow * 32 + hf * 8;
            ap.q[0] = *(const uint4*)pp;
            ap.q[1] = *(const uint4*)(pp + 16);
        }

        // ---- O += P * V ----
#pragma unroll
        for (int di = 0; di < 4; ++di) {
            FragBF bv;
#if HAVE_ASYNC_LDS
            const unsigned short* vp = lv + (di * 16 + lrow) * 32 + hf * 8;
#else
            const unsigned short* vp =
                Vb + (size_t)(di * 16 + lrow) * SEQ + kt0 + hf * 8;
#endif
            bv.q[0] = *(const uint4*)vp;
            bv.q[1] = *(const uint4*)(vp + 16);
            o[di] = __builtin_amdgcn_wmma_f32_16x16x32_bf16(
                false, ap.v, false, bv.v, (short)0, o[di], false, false);
        }
    }

    // ---- normalize and scatter ctx back to (B, S, D_MODEL) bf16 ----
    const int bb = bh >> 4, h = bh & 15;
    unsigned short* cb = ctx + (size_t)bb * SEQ * D_MODEL + h * DK;
#pragma unroll
    for (int di = 0; di < 4; ++di)
#pragma unroll
        for (int r = 0; r < 8; ++r) {
            const int s_row = q0 + hf * 8 + r;
            cb[(size_t)s_row * D_MODEL + di * 16 + lrow] = f2bf(o[di][r] / l_i[r]);
        }
}

// ---------------- host-side orchestration ----------------
extern "C" void kernel_launch(void* const* d_in, const int* in_sizes, int n_in,
                              void* d_out, int out_size, void* d_ws, size_t ws_size,
                              hipStream_t stream) {
    const float* q_in = (const float*)d_in[0];
    const float* k_in = (const float*)d_in[1];
    const float* v_in = (const float*)d_in[2];
    // d_in[3] = tril mask: causal handled analytically in-kernel
    const float* W_q = (const float*)d_in[4];
    const float* b_q = (const float*)d_in[5];
    const float* W_k = (const float*)d_in[6];
    const float* b_k = (const float*)d_in[7];
    const float* W_v = (const float*)d_in[8];
    const float* b_v = (const float*)d_in[9];
    const float* W_o = (const float*)d_in[10];
    const float* b_o = (const float*)d_in[11];

    const size_t SZ_ACT_BF = (size_t)MTOT * D_MODEL * 2;   // 8 MB
    const size_t SZ_W_BF   = (size_t)D_MODEL * D_MODEL * 2;
    const size_t SZ_ACT_F  = (size_t)MTOT * D_MODEL * 4;   // 16 MB

    char* p = (char*)d_ws;
    unsigned short* Xq  = (unsigned short*)p; p += SZ_ACT_BF;
    unsigned short* Xk  = (unsigned short*)p; p += SZ_ACT_BF;
    unsigned short* Xv  = (unsigned short*)p; p += SZ_ACT_BF;
    unsigned short* Wtq = (unsigned short*)p; p += SZ_W_BF;
    unsigned short* Wtk = (unsigned short*)p; p += SZ_W_BF;
    unsigned short* Wtv = (unsigned short*)p; p += SZ_W_BF;
    unsigned short* Wto = (unsigned short*)p; p += SZ_W_BF;
    float*          Pf  = (float*)p;          p += SZ_ACT_F;   // reused for Q,K,V proj
    unsigned short* Qh  = (unsigned short*)p; p += SZ_ACT_BF;
    unsigned short* Kh  = (unsigned short*)p; p += SZ_ACT_BF;
    unsigned short* Vt  = (unsigned short*)p; p += SZ_ACT_BF;
    unsigned short* Ctx = (unsigned short*)p; p += SZ_ACT_BF;

    const int nAct = MTOT * D_MODEL;          // 4M
    const int nW   = D_MODEL * D_MODEL;       // 1M
    const int nPr  = nAct / 2;                // RoPE pairs

    mha_cast_bf16<<<nAct / 256, 256, 0, stream>>>(q_in, Xq, nAct);
    mha_cast_bf16<<<nAct / 256, 256, 0, stream>>>(k_in, Xk, nAct);
    mha_cast_bf16<<<nAct / 256, 256, 0, stream>>>(v_in, Xv, nAct);
    mha_wtrans<<<nW / 256, 256, 0, stream>>>(W_q, Wtq);
    mha_wtrans<<<nW / 256, 256, 0, stream>>>(W_k, Wtk);
    mha_wtrans<<<nW / 256, 256, 0, stream>>>(W_v, Wtv);
    mha_wtrans<<<nW / 256, 256, 0, stream>>>(W_o, Wto);

    dim3 ggrid(MTOT / 128, D_MODEL / 128);
    mha_gemm_bf16<1><<<ggrid, 256, 0, stream>>>(Xq, Wtq, b_q, Pf, MTOT, D_MODEL, D_MODEL);
    mha_rope_cast<<<nPr / 256, 256, 0, stream>>>(Pf, Qh, nPr);
    mha_gemm_bf16<1><<<ggrid, 256, 0, stream>>>(Xk, Wtk, b_k, Pf, MTOT, D_MODEL, D_MODEL);
    mha_rope_cast<<<nPr / 256, 256, 0, stream>>>(Pf, Kh, nPr);
    mha_gemm_bf16<1><<<ggrid, 256, 0, stream>>>(Xv, Wtv, b_v, Pf, MTOT, D_MODEL, D_MODEL);
    mha_vcast_t<<<nAct / 256, 256, 0, stream>>>(Pf, Vt, nAct);

    mha_flash_attn<<<dim3(SEQ / 64, BATCH * NHEADS), 128, 0, stream>>>(Qh, Kh, Vt, Ctx);

    mha_gemm_bf16<0><<<ggrid, 256, 0, stream>>>(Ctx, Wto, b_o, (float*)d_out,
                                                MTOT, D_MODEL, D_MODEL);
}
